// DP_3427383902576
// MI455X (gfx1250) — matrix operand order repaired and checked by reading, not compiled, not run on hardware
//
#include <hip/hip_runtime.h>
#include <math.h>

typedef __attribute__((ext_vector_type(16))) _Float16 v16h;
typedef __attribute__((ext_vector_type(8)))  _Float16 v8h;
typedef __attribute__((ext_vector_type(2)))  _Float16 v2h;
typedef __attribute__((ext_vector_type(8)))  float    v8f;

#define DEV_INLINE __device__ __forceinline__

namespace {
constexpr int kNAtoms = 2048;   // NTYPES * NA
constexpr int kB      = 8;
constexpr int kE      = 25;     // embedding width
constexpr int kF      = 50;     // fit hidden width
constexpr int kLde    = 32;     // LDS row stride (elems), zero-padded for b128 frag loads

// ---- shared memory byte offsets (phase-overlaid) ----
// overlay region [0, 50176):
constexpr int SM_AH1  = 0;              // f16 [112][32]  7168 B
constexpr int SM_AH2  = 7168;
constexpr int SM_AG   = 14336;          // -> 21504
constexpr int SM_DP   = 21504;          // f32 [112][32] 14336 B
constexpr int SM_DO   = 35840;          // f32 [112][32] -> 50176
// fit-phase overlay of the same region:
constexpr int SM_DR16 = 0;              // f16 [16][416] 13312 B (zero-padded)
constexpr int SM_FH   = 13312;          // f32 [3][16][64] 12288 B (padding written as 0)
constexpr int SM_DPF0 = 25600;          // f16 [16][64] 2048 B
constexpr int SM_DPF1 = 27648;          // f16 [16][64] -> 29696
constexpr int SM_FC   = 29696;          // f32 201: Bf1,Bf2,Bf3,Wout,bout (fit phases only)
constexpr int SM_DDR  = 0;              // f32 [16][400] 25600 B (over dead DR16+FH)
// persistent regions:
constexpr int SM_SCAT = 50176;          // f32 1600
constexpr int SM_DSC  = 56576;          // f32 1600
constexpr int SM_BLK  = 62976;          // f32 400 (one atom's blk / dE stash)
constexpr int SM_EC   = 64576;          // f32 200: both emb nets' W1,B1,B2,B3
constexpr int SM_SIZE = 65376;          // < 64 KB
constexpr int kRegionF32 = 50176 / 4;   // floats to zero in overlay region
} // namespace

DEV_INLINE float fast_tanh(float x) {
  // tanh(x) = 1 - 2/(exp2(2*log2e*x) + 1); saturates correctly at +-1, branch-free
  const float e = __builtin_amdgcn_exp2f(x * 2.88539008177f);
  return 1.f - 2.f * __builtin_amdgcn_rcpf(e + 1.f);
}

DEV_INLINE v8f wmma_f16(v16h a, v16h b, v8f c) {
  return __builtin_amdgcn_wmma_f32_16x16x32_f16(false, a, false, b, (short)0, c,
                                                false, false);
}

// A fragment (16x32 f16) from zero-padded LDS f16, row stride ldk (mult of 16).
// Per lane: two contiguous 16-byte chunks -> unconditional ds_load_b128 x2.
DEV_INLINE v16h a_frag_h(const _Float16* act, int ldk, int row0, int k0) {
  const int lane = threadIdx.x & 31;
  const _Float16* p = act + (row0 + (lane & 15)) * ldk + k0 + ((lane & 16) >> 1);
  const v8h lo = *(const v8h*)p;
  const v8h hi = *(const v8h*)(p + 16);
  return __builtin_shufflevector(lo, hi, 0, 1, 2, 3, 4, 5, 6, 7, 8, 9, 10, 11,
                                 12, 13, 14, 15);
}

// Same, from zero-padded f32 LDS (converted to f16 on the fly).
DEV_INLINE v16h a_frag_f(const float* act, int ldk, int row0, int k0) {
  const int lane = threadIdx.x & 31;
  const float* p = act + (row0 + (lane & 15)) * ldk + k0 + ((lane & 16) >> 1);
  v16h a;
#pragma unroll
  for (int h = 0; h < 8; ++h) {
    a[h] = (_Float16)p[h];
    a[h + 8] = (_Float16)p[h + 16];
  }
  return a;
}

// B fragment (32x16 f16) from global f32 W[k][n]; clamp+mask, all loads unconditional.
DEV_INLINE v16h b_frag(const float* W, int ld, int k0, int kvalid, int n0,
                       int nvalid) {
  const int lane = threadIdx.x & 31;
  const int n = n0 + (lane & 15);
  const int nc = (n < nvalid) ? n : 0;
  const float nm = (n < nvalid) ? 1.f : 0.f;
  const int kb = k0 + ((lane & 16) ? 8 : 0);
  v16h b;
#pragma unroll
  for (int h = 0; h < 16; ++h) {
    const int k = kb + ((h < 8) ? h : h + 8);
    const int kc = (k < kvalid) ? k : 0;
    const float m = (k < kvalid) ? nm : 0.f;
    b[h] = (_Float16)(W[kc * ld + nc] * m);
  }
  return b;
}

// Transposed-weight B fragment: B[k][n] = W[n][k] (for dX = dH @ W^T).
DEV_INLINE v16h bt_frag(const float* W, int ld, int k0, int kvalid, int n0,
                        int nvalid) {
  const int lane = threadIdx.x & 31;
  const int n = n0 + (lane & 15);
  const int nc = (n < nvalid) ? n : 0;
  const float nm = (n < nvalid) ? 1.f : 0.f;
  const int kb = k0 + ((lane & 16) ? 8 : 0);
  v16h b;
#pragma unroll
  for (int h = 0; h < 16; ++h) {
    const int k = kb + ((h < 8) ? h : h + 8);
    const int kc = (k < kvalid) ? k : 0;
    const float m = (k < kvalid) ? nm : 0.f;
    b[h] = (_Float16)(W[nc * ld + kc] * m);
  }
  return b;
}

struct DPParams {
  const float* Ri;     // (B, 2048, 200, 4)
  const float* dfeat;  // (B, 2048, 800, 3)
  const float* embW1[4]; const float* embB1[4];
  const float* embW2[4]; const float* embB2[4];
  const float* embW3[4]; const float* embB3[4];
  const float* fitW[2][3]; const float* fitB[2][3];
  const float* Wout[2]; const float* bout[2];
  float* out;          // [8 Etot][8*2048 Ei][8*2048*3 F]
};

// Embedding net forward for 100 neighbors (one atom, one type-block).
// ECn: LDS consts [0..24]=W1 [25..49]=B1 [50..74]=B2 [75..99]=B3.
// h1 = tanh(s*W1+b1); h2 = tanh(h1@W2+b2)+h1; G = tanh(h2@W3+b3)+h2
DEV_INLINE void emb_forward(const float* BLK, const float* ECn, const float* W2,
                            const float* W3, _Float16* AH1, _Float16* AH2,
                            _Float16* AG) {
  const int tid = threadIdx.x, wave = tid >> 5, lane = tid & 31;
  for (int k = tid; k < 100; k += 256) {
    const float s = BLK[k * 4];
    _Float16* dstp = AH1 + k * kLde;
#pragma unroll
    for (int j = 0; j < 24; j += 2) {  // packed f16 pair stores
      v2h pr;
      pr[0] = (_Float16)fast_tanh(s * ECn[j] + ECn[25 + j]);
      pr[1] = (_Float16)fast_tanh(s * ECn[j + 1] + ECn[25 + j + 1]);
      *(v2h*)(dstp + j) = pr;
    }
    v2h pr;
    pr[0] = (_Float16)fast_tanh(s * ECn[24] + ECn[49]);
    pr[1] = (_Float16)0.f;
    *(v2h*)(dstp + 24) = pr;
  }
  __syncthreads();
  const float* Ws[2] = {W2, W3};
  _Float16* src = AH1;
  _Float16* dst = AH2;
  for (int L = 0; L < 2; ++L) {
    for (int u = wave; u < 14; u += 8) {  // 7 M-tiles x 2 N-tiles
      const int mt = u % 7, nt = u / 7;
      v16h a = a_frag_h(src, kLde, mt * 16, 0);
      v16h bf = b_frag(Ws[L], kE, 0, kE, nt * 16, kE);
      v8f c = {};
      c = wmma_f16(a, bf, c);
      const int cm0 = (lane & 16) ? 8 : 0, cn = lane & 15;
      const int col = nt * 16 + cn;
      const float mcol = (col < kE) ? 1.f : 0.f;
      const float bias = ECn[50 + L * 25 + ((col < kE) ? col : kE - 1)];  // LDS
#pragma unroll
      for (int r = 0; r < 8; ++r) {  // branch-free masked epilogue
        const int row = mt * 16 + r + cm0;
        const float hv = fast_tanh(c[r] + bias);
        const float res = (float)src[row * kLde + col];
        const float m = (row < 100) ? mcol : 0.f;
        dst[row * kLde + col] = (_Float16)((hv + res) * m);
      }
    }
    __syncthreads();
    src = dst;
    dst = AG;
  }
}

__global__ __launch_bounds__(256) void dp_fused(DPParams P) {
  __shared__ __align__(16) unsigned char smem[SM_SIZE];
  _Float16* AH1 = reinterpret_cast<_Float16*>(smem + SM_AH1);
  _Float16* AH2 = reinterpret_cast<_Float16*>(smem + SM_AH2);
  _Float16* AG  = reinterpret_cast<_Float16*>(smem + SM_AG);
  float*    DPb = reinterpret_cast<float*>(smem + SM_DP);
  float*    DOb = reinterpret_cast<float*>(smem + SM_DO);
  _Float16* DR16 = reinterpret_cast<_Float16*>(smem + SM_DR16);
  float*    FHb  = reinterpret_cast<float*>(smem + SM_FH);  // [L*1024 + a*64 + j]
  _Float16* DPF0 = reinterpret_cast<_Float16*>(smem + SM_DPF0);
  _Float16* DPF1 = reinterpret_cast<_Float16*>(smem + SM_DPF1);
  float*    FC   = reinterpret_cast<float*>(smem + SM_FC);
  float*    DDR  = reinterpret_cast<float*>(smem + SM_DDR);
  float*    SCAT = reinterpret_cast<float*>(smem + SM_SCAT);
  float*    DSC  = reinterpret_cast<float*>(smem + SM_DSC);
  float*    BLK  = reinterpret_cast<float*>(smem + SM_BLK);
  float*    EC   = reinterpret_cast<float*>(smem + SM_EC);
  float*    FACC = SCAT;  // phase-6 overlay (scat dead by then)
  float*    REG  = reinterpret_cast<float*>(smem);  // overlay region as f32

  const int tid = threadIdx.x, wave = tid >> 5, lane = tid & 31;
  const int b = blockIdx.x >> 7;
  const int atom0 = (blockIdx.x & 127) << 4;
  const int it = atom0 >> 10;

  for (int i = tid; i < kRegionF32; i += 256) REG[i] = 0.f;  // zero padding
  for (int i = tid; i < 1600; i += 256) SCAT[i] = 0.f;
  for (int i = tid; i < 200; i += 256) {  // stage emb consts to LDS (once)
    const int jt = i / 100, off = i % 100, f = off / 25, j = off % 25;
    const int net = it * 2 + jt;
    const float* sp = (f == 0)   ? P.embW1[net]
                      : (f == 1) ? P.embB1[net]
                      : (f == 2) ? P.embB2[net]
                                 : P.embB3[net];
    EC[i] = sp[j];
  }
  __syncthreads();

  // ---------------- Phase 1: embedding forward + scat ----------------
  for (int a = 0; a < 16; ++a) {
    const int atom = atom0 + a;
    for (int jt = 0; jt < 2; ++jt) {
      const int net = it * 2 + jt;
      const float* rb = P.Ri + ((size_t)(b * kNAtoms + atom) * 200 + jt * 100) * 4;
      for (int i = tid; i < 100; i += 256)  // b128 coalesced stage
        ((float4*)BLK)[i] = ((const float4*)rb)[i];
      __syncthreads();
      emb_forward(BLK, EC + jt * 100, P.embW2[net], P.embW3[net], AH1, AH2, AG);
      // t[d][e] += sum_k blk[k][d] * G[k][e]
      for (int idx = tid; idx < 100; idx += 256) {
        const int d = idx / kE, e = idx % kE;
        float s = 0.f;
        for (int k = 0; k < 100; ++k)
          s += BLK[k * 4 + d] * (float)AG[k * kLde + e];
        SCAT[a * 100 + d * kE + e] += s;
      }
      __syncthreads();
    }
  }
  // ---------------- Phase 2: scale + DR + stage fit consts ----------------
  for (int i = tid; i < 1600; i += 256) SCAT[i] *= (1.f / 200.f);
  __syncthreads();
  for (int idx = tid; idx < 16 * 416; idx += 256) {
    const int a = idx / 416, q = idx % 416;
    float v = 0.f;
    if (q < 400) {
      const int e = q >> 4, m = q & 15;
      const float* sc = SCAT + a * 100;
      for (int d = 0; d < 4; ++d) v += sc[d * kE + e] * sc[d * kE + m];
    }
    DR16[idx] = (_Float16)v;
  }
  if (tid < kF) {  // fit consts -> LDS (dead hole in fit overlay)
    FC[tid] = P.fitB[it][0][tid];
    FC[50 + tid] = P.fitB[it][1][tid];
    FC[100 + tid] = P.fitB[it][2][tid];
    FC[150 + tid] = P.Wout[it][tid];
  }
  if (tid == 0) FC[200] = P.bout[it][0];
  __syncthreads();

  // ---------------- Phase 3: fit forward (M = the 16 atoms) ----------------
  const float* Wf1 = P.fitW[it][0];
  const float* Wf2 = P.fitW[it][1];
  const float* Wf3 = P.fitW[it][2];

  if (wave < 4) {  // F1: 400 -> 50 (4 N-tiles, 13 K-chunks)
    const int nt = wave;
    v8f c = {};
    for (int kc = 0; kc < 13; ++kc) {
      v16h a = a_frag_h(DR16, 416, 0, kc * 32);
      v16h bf = b_frag(Wf1, kF, kc * 32, 400, nt * 16, kF);
      c = wmma_f16(a, bf, c);
    }
    const int cm0 = (lane & 16) ? 8 : 0, cn = lane & 15;
    const int col = nt * 16 + cn;
    const float mcol = (col < kF) ? 1.f : 0.f;
    const float bias = FC[(col < kF) ? col : kF - 1];
#pragma unroll
    for (int r = 0; r < 8; ++r)
      FHb[(r + cm0) * 64 + col] = fast_tanh(c[r] + bias) * mcol;
  }
  __syncthreads();
  for (int L = 1; L <= 2; ++L) {  // 50 -> 50 + residual
    const float* W = (L == 1) ? Wf2 : Wf3;
    if (wave < 4) {
      const int nt = wave;
      v8f c = {};
      for (int kc = 0; kc < 2; ++kc) {
        v16h a = a_frag_f(FHb + (L - 1) * 1024, 64, 0, kc * 32);
        v16h bf = b_frag(W, kF, kc * 32, kF, nt * 16, kF);
        c = wmma_f16(a, bf, c);
      }
      const int cm0 = (lane & 16) ? 8 : 0, cn = lane & 15;
      const int col = nt * 16 + cn;
      const float mcol = (col < kF) ? 1.f : 0.f;
      const float bias = FC[L * 50 + ((col < kF) ? col : kF - 1)];
#pragma unroll
      for (int r = 0; r < 8; ++r) {
        const int row = r + cm0;
        const float hv = fast_tanh(c[r] + bias);
        FHb[L * 1024 + row * 64 + col] =
            (hv + FHb[(L - 1) * 1024 + row * 64 + col]) * mcol;
      }
    }
    __syncthreads();
  }
  if (tid < 16) {  // Ei
    float e = FC[200];
    for (int j = 0; j < kF; ++j) e += FHb[2048 + tid * 64 + j] * FC[150 + j];
    P.out[8 + b * kNAtoms + atom0 + tid] = e;
  }
  __syncthreads();

  // ---------------- Phase 4: fit backward -> dDR ----------------
  for (int idx = tid; idx < 16 * 64; idx += 256) {  // dpre3 (full 64 cols, pad=0)
    const int a = idx >> 6, j = idx & 63;
    const float m = (j < kF) ? 1.f : 0.f;
    const float t3 = FHb[2048 + a * 64 + j] - FHb[1024 + a * 64 + j];
    DPF0[idx] = (_Float16)(FC[150 + ((j < kF) ? j : kF - 1)] * (1.f - t3 * t3) * m);
  }
  __syncthreads();
  {
    const int nt = wave, cn = lane & 15, cm0 = (lane & 16) ? 8 : 0;
    const int col = nt * 16 + cn;
    float doutv[8];
    const float w0 = (wave < 4 && col < kF) ? FC[150 + col] : 0.f;
#pragma unroll
    for (int r = 0; r < 8; ++r) doutv[r] = w0;
    for (int step = 0; step < 2; ++step) {  // dpre3@Wf3^T, then dpre2@Wf2^T
      const float* W = (step == 0) ? Wf3 : Wf2;
      _Float16* Ain = (step == 0) ? DPF0 : DPF1;
      _Float16* Aout = (step == 0) ? DPF1 : DPF0;
      if (wave < 4) {
        v8f c = {};
        for (int kc = 0; kc < 2; ++kc) {
          v16h a = a_frag_h(Ain, 64, 0, kc * 32);
          v16h bf = bt_frag(W, kF, kc * 32, kF, nt * 16, kF);
          c = wmma_f16(a, bf, c);
        }
#pragma unroll
        for (int r = 0; r < 8; ++r) {  // padding lanes produce exact 0
          const int row = r + cm0;
          const float dn = c[r] + doutv[r];  // + residual skip
          doutv[r] = dn;
          const float t = (step == 0)
                              ? (FHb[1024 + row * 64 + col] - FHb[row * 64 + col])
                              : FHb[row * 64 + col];
          Aout[row * 64 + col] = (_Float16)(dn * (1.f - t * t));
        }
      }
      __syncthreads();
    }
  }
  for (int nt = wave; nt < 25; nt += 8) {  // dDR = dpre1 @ Wf1^T (25 N-tiles)
    v8f c = {};
    for (int kc = 0; kc < 2; ++kc) {
      v16h a = a_frag_h(DPF0, 64, 0, kc * 32);
      v16h bf = bt_frag(Wf1, kF, kc * 32, kF, nt * 16, 400);
      c = wmma_f16(a, bf, c);
    }
    const int cm0 = (lane & 16) ? 8 : 0, cn = lane & 15;
#pragma unroll
    for (int r = 0; r < 8; ++r)  // col = nt*16+cn <= 399 always
      DDR[(r + cm0) * 400 + nt * 16 + cn] = c[r];
  }
  __syncthreads();

  // ---------------- Phase 5: dscat (incl. 1/200 for dt) ----------------
  for (int idx = tid; idx < 1600; idx += 256) {
    const int a = idx / 100, rdm = idx % 100, d = rdm / kE, e = rdm % kE;
    const float* sc = SCAT + a * 100 + d * kE;
    const float* dr = DDR + a * 400;
    float v = 0.f;
    for (int m = 0; m < 16; ++m) v += dr[e * 16 + m] * sc[m];
    if (e < 16)
      for (int e2 = 0; e2 < kE; ++e2) v += dr[e2 * 16 + e] * sc[e2];
    DSC[idx] = v * (1.f / 200.f);
  }
  __syncthreads();
  if (tid < 48) FACC[tid] = 0.f;  // overlays SCAT (now dead)
  for (int i = tid; i < kRegionF32; i += 256) REG[i] = 0.f;  // re-zero padding
  __syncthreads();

  // ---------------- Phase 6: embedding backward (recompute) + forces ----------------
  for (int a = 0; a < 16; ++a) {
    const int atom = atom0 + a;
    const float* dsc = DSC + a * 100;  // [d*25+e]
    for (int jt = 0; jt < 2; ++jt) {
      const int net = it * 2 + jt;
      const float* ECn = EC + jt * 100;
      const float* rb = P.Ri + ((size_t)(b * kNAtoms + atom) * 200 + jt * 100) * 4;
      for (int i = tid; i < 100; i += 256)
        ((float4*)BLK)[i] = ((const float4*)rb)[i];
      __syncthreads();
      emb_forward(BLK, ECn, P.embW2[net], P.embW3[net], AH1, AH2, AG);
      // dG[k][e] = sum_d dt[d][e]*blk[k][d] ; dpre3 = dG*(1-t3^2)
      for (int idx = tid; idx < 2500; idx += 256) {
        const int k = idx / kE, e = idx % kE;
        const float g = dsc[e] * BLK[k * 4] + dsc[25 + e] * BLK[k * 4 + 1] +
                        dsc[50 + e] * BLK[k * 4 + 2] + dsc[75 + e] * BLK[k * 4 + 3];
        DOb[k * kLde + e] = g;
        const float t3 = (float)AG[k * kLde + e] - (float)AH2[k * kLde + e];
        DPb[k * kLde + e] = g * (1.f - t3 * t3);
      }
      __syncthreads();
      for (int step = 0; step < 2; ++step) {  // @W3^T then @W2^T, + residual
        const float* W = (step == 0) ? P.embW3[net] : P.embW2[net];
        for (int u = wave; u < 14; u += 8) {
          const int mt = u % 7, nt = u / 7;
          v16h af = a_frag_f(DPb, kLde, mt * 16, 0);
          v16h bf = bt_frag(W, kE, 0, kE, nt * 16, kE);
          v8f c = {};
          c = wmma_f16(af, bf, c);
          const int cm0 = (lane & 16) ? 8 : 0, cn = lane & 15;
          const int col = nt * 16 + cn;
#pragma unroll
          for (int r = 0; r < 8; ++r) {  // unconditional RMW (c==0 in padding)
            const int row = mt * 16 + r + cm0;
            DOb[row * kLde + col] = c[r] + DOb[row * kLde + col];
          }
        }
        __syncthreads();
        if (step == 0) {  // dpre2
          for (int idx = tid; idx < 2500; idx += 256) {
            const int k = idx / kE, e = idx % kE;
            const float t2 = (float)AH2[k * kLde + e] - (float)AH1[k * kLde + e];
            DPb[k * kLde + e] = DOb[k * kLde + e] * (1.f - t2 * t2);
          }
          __syncthreads();
        }
      }
      // dE[k*4+d] = dblk[k][d] (+ ds[k] if d==0); stash into BLK (blk now dead)
      for (int idx = tid; idx < 400; idx += 256) {
        const int k = idx >> 2, d = idx & 3;
        float dE = 0.f;
        for (int e = 0; e < kE; ++e)
          dE += dsc[d * kE + e] * (float)AG[k * kLde + e];
        if (d == 0) {
          float ds = 0.f;
          for (int j = 0; j < kE; ++j) {
            const float h1 = (float)AH1[k * kLde + j];
            ds += DOb[k * kLde + j] * (1.f - h1 * h1) * ECn[j];
          }
          dE += ds;
        }
        BLK[idx] = dE;
      }
      __syncthreads();
      // force contraction, b128-coalesced over dfeat (the 157 MB stream)
      {
        const float* dfr =
            P.dfeat + ((size_t)(b * kNAtoms + atom) * 800 + jt * 400) * 3;
        for (int i = tid; i < 300; i += 256) {
          const float4 dv = ((const float4*)dfr)[i];
          const int i3 = i * 4;
          float4 o;
          o.x = -BLK[(i3 + 0) / 3] * dv.x;
          o.y = -BLK[(i3 + 1) / 3] * dv.y;
          o.z = -BLK[(i3 + 2) / 3] * dv.z;
          o.w = -BLK[(i3 + 3) / 3] * dv.w;
          ((float4*)DOb)[i] = o;
        }
      }
      __syncthreads();
      float* FPART = DOb + 1280;
      if (tid < 48) {  // fixed-order partial sums (deterministic)
        const int g = tid / 3, x = tid % 3;
        float s = 0.f;
        for (int q = 0; q < 25; ++q) s += DOb[(g * 25 + q) * 3 + x];
        FPART[tid] = s;
      }
      __syncthreads();
      if (tid < 3) {
        float s = 0.f;
        for (int g = 0; g < 16; ++g) s += FPART[g * 3 + tid];
        FACC[a * 3 + tid] += s;
      }
      __syncthreads();
    }
  }
  if (tid < 48) {
    const int a = tid / 3, x = tid % 3;
    P.out[8 + kB * kNAtoms + (size_t)(b * kNAtoms + atom0 + a) * 3 + x] =
        FACC[tid];
  }
}

__global__ __launch_bounds__(256) void etot_reduce(float* out) {
  __shared__ float red[256];
  const int b = blockIdx.x;
  const float* Ei = out + 8 + b * kNAtoms;
  float s = 0.f;
  for (int i = threadIdx.x; i < kNAtoms; i += 256) s += Ei[i];
  red[threadIdx.x] = s;
  __syncthreads();
  for (int off = 128; off > 0; off >>= 1) {
    if (threadIdx.x < off) red[threadIdx.x] += red[threadIdx.x + off];
    __syncthreads();
  }
  if (threadIdx.x == 0) out[b] = red[0];
}

extern "C" void kernel_launch(void* const* d_in, const int* in_sizes, int n_in,
                              void* d_out, int out_size, void* d_ws,
                              size_t ws_size, hipStream_t stream) {
  (void)in_sizes; (void)n_in; (void)out_size; (void)d_ws; (void)ws_size;
  DPParams P;
  P.Ri = (const float*)d_in[0];
  P.dfeat = (const float*)d_in[1];
  // d_in[2..5]: list_neigh / natoms_img / atom_type / ImageDR (unused by reference math)
  for (int net = 0; net < 4; ++net) {
    P.embW1[net] = (const float*)d_in[6 + net * 6 + 0];
    P.embB1[net] = (const float*)d_in[6 + net * 6 + 1];
    P.embW2[net] = (const float*)d_in[6 + net * 6 + 2];
    P.embB2[net] = (const float*)d_in[6 + net * 6 + 3];
    P.embW3[net] = (const float*)d_in[6 + net * 6 + 4];
    P.embB3[net] = (const float*)d_in[6 + net * 6 + 5];
  }
  for (int t = 0; t < 2; ++t) {
    for (int l = 0; l < 3; ++l) {
      P.fitW[t][l] = (const float*)d_in[30 + t * 8 + l * 2];
      P.fitB[t][l] = (const float*)d_in[30 + t * 8 + l * 2 + 1];
    }
    P.Wout[t] = (const float*)d_in[30 + t * 8 + 6];
    P.bout[t] = (const float*)d_in[30 + t * 8 + 7];
  }
  P.out = (float*)d_out;

  dp_fused<<<(kB * kNAtoms) / 16, 256, 0, stream>>>(P);
  etot_reduce<<<kB, 256, 0, stream>>>((float*)d_out);
}
